// TransductiveGCN_8254927143023
// MI455X (gfx1250) — matrix operand order repaired and checked by reading, not compiled, not run on hardware
//
#include <hip/hip_runtime.h>
#include <math.h>

typedef __attribute__((ext_vector_type(8)))  unsigned short v8us;
typedef __attribute__((ext_vector_type(16))) unsigned short v16us;
typedef __attribute__((ext_vector_type(16))) __bf16         v16bf;
typedef __attribute__((ext_vector_type(8)))  float          v8f;

__device__ __forceinline__ unsigned short f2bf(float x) {
    unsigned u = __float_as_uint(x);
    return (unsigned short)((u + 0x7FFFu + ((u >> 16) & 1u)) >> 16);  // RNE
}

enum { EPI_PLAIN = 0, EPI_BNRELU = 1, EPI_OUT = 2 };

// ---------------------------------------------------------------------------
// WMMA bf16 GEMM, N-blocked with LDS-staged B:
//   A  : bf16 [M x KP]   row-major, K zero-padded to KP (mult of 32); M%16==0
//   Bt : bf16 [NP x KP]  B transposed (column-major of B), zero-padded
// NP = NT*16 = full output width; one wave owns a 16-row strip and all NT
// column tiles (NT accumulators, NT WMMAs per A fragment). Whole Bt panel
// (<=16KB) staged in LDS once per block; fragments via ds_load_b128.
// Epilogue is a compile-time variant:
//   EPI_PLAIN : C f32 [M x NP]                        (GCN linear)
//   EPI_BNRELU: Cb bf16 [M x NP] = relu(bn(acc+bias)) (MLP hidden)
//   EPI_OUT   : C f32 [M x Ncols] = acc+bias, col-guarded (head)
// ---------------------------------------------------------------------------
template <int KP, int NT, int EPI>
__global__ void wmma_gemm_kernel(const unsigned short* __restrict__ A,
                                 const unsigned short* __restrict__ Bt,
                                 const float* __restrict__ bias,
                                 const float* __restrict__ bn_g,
                                 const float* __restrict__ bn_b,
                                 float* __restrict__ C,
                                 unsigned short* __restrict__ Cb,
                                 int M, int Ncols) {
    constexpr int NP = NT * 16;
    __shared__ unsigned short sB[NP * KP];

    const int wave   = threadIdx.x >> 5;
    const int lane   = threadIdx.x & 31;
    const int tilesM = M >> 4;                    // M % 16 == 0 by construction
    int tm = blockIdx.x * 8 + wave;
    const bool live = (tm < tilesM);
    if (!live) tm = tilesM - 1;                   // clamp; skip stores later

    // cooperative stage of the whole Bt panel into LDS (b128 copies)
    #pragma unroll
    for (int idx = threadIdx.x; idx < NP * KP / 8; idx += 256) {
        *(v8us*)(sB + idx * 8) = *(const v8us*)(Bt + idx * 8);
    }
    __syncthreads();

    const int half = lane >> 4;
    const int mn   = lane & 15;
    const unsigned short* Ap = A + (long long)(tm * 16 + mn) * KP;

    v8f acc[NT] = {};
    #pragma unroll
    for (int k0 = 0; k0 < KP; k0 += 32) {
        // A 16x32 frag: per-lane K chunks [k0+8h, +8) and [k0+16+8h, +8)
        v8us a0 = *(const v8us*)(Ap + k0 + 8 * half);
        v8us a1 = *(const v8us*)(Ap + k0 + 16 + 8 * half);
        v16us au = __builtin_shufflevector(a0, a1, 0, 1, 2, 3, 4, 5, 6, 7,
                                                   8, 9, 10, 11, 12, 13, 14, 15);
        #pragma unroll
        for (int t = 0; t < NT; ++t) {
            // B 32x16 frag: column (t*16+mn), K chunks [k0+8h, +8) and [k0+16+8h, +8) from LDS
            v8us b0 = *(const v8us*)(sB + (t * 16 + mn) * KP + k0 + 8 * half);
            v8us b1 = *(const v8us*)(sB + (t * 16 + mn) * KP + k0 + 16 + 8 * half);
            v16us bu = __builtin_shufflevector(b0, b1, 0, 1, 2, 3, 4, 5, 6, 7,
                                                       8, 9, 10, 11, 12, 13, 14, 15);
            acc[t] = __builtin_amdgcn_wmma_f32_16x16x32_bf16(
                         false, __builtin_bit_cast(v16bf, au),
                         false, __builtin_bit_cast(v16bf, bu),
                         (short)0, acc[t], false, false);
        }
    }
    if (!live) return;                            // wave-uniform

    #pragma unroll
    for (int t = 0; t < NT; ++t) {
        const int col = t * 16 + mn;
        const int row0 = tm * 16 + 8 * half;      // C: VGPR r -> M = r + 8*half
        if (EPI == EPI_PLAIN) {
            #pragma unroll
            for (int r = 0; r < 8; ++r)
                C[(long long)(row0 + r) * NP + col] = acc[t][r];
        } else if (EPI == EPI_BNRELU) {
            const float b  = bias[col];
            const float sc = bn_g[col] * rsqrtf(1.0f + 1e-5f);
            const float sh = bn_b[col];
            #pragma unroll
            for (int r = 0; r < 8; ++r) {
                float v = fmaxf((acc[t][r] + b) * sc + sh, 0.0f);
                Cb[(long long)(row0 + r) * NP + col] = f2bf(v);
            }
        } else {                                  // EPI_OUT
            if (col < Ncols) {
                const float b = bias[col];
                #pragma unroll
                for (int r = 0; r < 8; ++r)
                    C[(long long)(row0 + r) * Ncols + col] = acc[t][r] + b;
            }
        }
    }
}

// ---------------------------------------------------------------------------
__global__ void fill_kernel(float* __restrict__ p, float v, long long n) {
    long long t = (long long)blockIdx.x * blockDim.x + threadIdx.x;
    if (t < n) p[t] = v;
}

// f32 [M x K] -> bf16 [M x KP], zero-padded in K
__global__ void pack_a_kernel(const float* __restrict__ A, unsigned short* __restrict__ Ab,
                              long long M, int K, int KP) {
    long long t = (long long)blockIdx.x * blockDim.x + threadIdx.x;
    if (t >= M * KP) return;
    long long row = t / KP;
    int k = (int)(t % KP);
    Ab[t] = f2bf(k < K ? A[row * K + k] : 0.0f);
}

// f32 B [K x N] -> bf16 Bt [NP x KP] (transposed), zero-padded
__global__ void pack_bt_kernel(const float* __restrict__ B, unsigned short* __restrict__ Bt,
                               int K, int N, int KP, int NP) {
    long long t = (long long)blockIdx.x * blockDim.x + threadIdx.x;
    if (t >= (long long)NP * KP) return;
    int n = (int)(t / KP);
    int k = (int)(t % KP);
    Bt[t] = f2bf((k < K && n < N) ? B[(long long)k * N + n] : 0.0f);
}

__global__ void deg_kernel(const int* __restrict__ dst, float* __restrict__ deg, int E) {
    long long t = (long long)blockIdx.x * blockDim.x + threadIdx.x;
    if (t < E) atomicAdd(&deg[dst[t]], 1.0f);
}

__global__ void rsqrt_kernel(const float* __restrict__ deg, float* __restrict__ dinv, int N) {
    long long t = (long long)blockIdx.x * blockDim.x + threadIdx.x;
    if (t < N) dinv[t] = rsqrtf(deg[t]);
}

// one thread per (edge, 4 features): AGG[dst] += H[src] * dinv[src]*dinv[dst]
__global__ void edge_agg_kernel(const int* __restrict__ src, const int* __restrict__ dst,
                                const float* __restrict__ dinv,
                                const float* __restrict__ H,
                                float* __restrict__ AGG, long long total) {
    long long t = (long long)blockIdx.x * blockDim.x + threadIdx.x;
    if (t >= total) return;                        // total = E*16, HID=64
    long long e = t >> 4;
    int fq = (int)(t & 15) * 4;
    int s = src[e], d = dst[e];
    float c = dinv[s] * dinv[d];
    const float4 h = *(const float4*)(H + (long long)s * 64 + fq);
    float* a = AGG + (long long)d * 64 + fq;
    atomicAdd(a + 0, h.x * c);
    atomicAdd(a + 1, h.y * c);
    atomicAdd(a + 2, h.z * c);
    atomicAdd(a + 3, h.w * c);
}

// X = relu(AGG + H/deg + b), written as f32 and bf16
__global__ void combine_kernel(const float* __restrict__ H, const float* __restrict__ AGG,
                               const float* __restrict__ deg, const float* __restrict__ b,
                               float* __restrict__ X, unsigned short* __restrict__ Xb,
                               long long total) {
    long long t = (long long)blockIdx.x * blockDim.x + threadIdx.x;
    if (t >= total) return;
    int f = (int)(t & 63);
    long long i = t >> 6;
    float v = AGG[t] + H[t] / deg[i] + b[f];
    v = fmaxf(v, 0.0f);
    X[t] = v;
    Xb[t] = f2bf(v);
}

// mean-sum into emb[:, 0:64], float max (h>=0) into emb[:, 64:128]
__global__ void pool_kernel(const float* __restrict__ X, const int* __restrict__ batch,
                            float* __restrict__ emb, float* __restrict__ counts,
                            long long total) {
    long long t = (long long)blockIdx.x * blockDim.x + threadIdx.x;
    if (t >= total) return;
    int f = (int)(t & 63);
    long long i = t >> 6;
    int g = batch[i];
    float h = X[t];
    atomicAdd(&emb[(long long)g * 128 + f], h);
    atomicMax((int*)&emb[(long long)g * 128 + 64 + f], __float_as_int(h));
    if (f == 0) atomicAdd(&counts[g], 1.0f);
}

__global__ void pool_fin_kernel(float* __restrict__ emb, const float* __restrict__ counts,
                                long long total) {
    long long t = (long long)blockIdx.x * blockDim.x + threadIdx.x;
    if (t >= total) return;                        // total = G*64
    int f = (int)(t & 63);
    long long g = t >> 6;
    emb[g * 128 + f] /= fmaxf(counts[g], 1.0f);
}

// ---------------------------------------------------------------------------
static inline long long cdivll(long long a, long long b) { return (a + b - 1) / b; }

extern "C" void kernel_launch(void* const* d_in, const int* in_sizes, int n_in,
                              void* d_out, int out_size, void* d_ws, size_t ws_size,
                              hipStream_t stream) {
    const float* x     = (const float*)d_in[0];
    const int*   ei    = (const int*)d_in[1];
    const int*   batch = (const int*)d_in[2];
    const float* W1    = (const float*)d_in[3];
    const float* b1    = (const float*)d_in[4];
    const float* W2    = (const float*)d_in[5];
    const float* b2    = (const float*)d_in[6];
    const float* W3    = (const float*)d_in[7];
    const float* b3    = (const float*)d_in[8];
    const float* cw1   = (const float*)d_in[9];
    const float* cb1   = (const float*)d_in[10];
    const float* g1    = (const float*)d_in[11];
    const float* be1   = (const float*)d_in[12];
    const float* cw2   = (const float*)d_in[13];
    const float* cb2   = (const float*)d_in[14];
    const float* g2    = (const float*)d_in[15];
    const float* be2   = (const float*)d_in[16];
    const float* cw3   = (const float*)d_in[17];
    const float* cb3   = (const float*)d_in[18];

    const int HID = in_sizes[4];                 // 64
    const int IN  = in_sizes[3] / HID;           // 36
    const int N   = in_sizes[0] / IN;            // 100000 (multiple of 16)
    const int E   = in_sizes[1] / 2;             // 1200000
    const int TGT = in_sizes[18];                // 5
    const int G   = out_size / TGT;              // 2000   (multiple of 16)
    const int H2  = HID / 2;                     // 32
    const int KPI = (IN + 31) & ~31;             // 64 (padded IN)

    const int* src = ei;
    const int* dst = ei + E;

    // --- workspace carve (256B-aligned chunks) ---
    char*  base = (char*)d_ws;
    size_t off  = 0;
    auto alloc = [&](size_t bytes) -> void* {
        void* p = base + off;
        off += (bytes + 255) & ~(size_t)255;
        return p;
    };
    float* deg    = (float*)alloc(sizeof(float) * N);
    float* dinv   = (float*)alloc(sizeof(float) * N);
    float* Hbuf   = (float*)alloc(sizeof(float) * (size_t)N * HID);
    float* AGG    = (float*)alloc(sizeof(float) * (size_t)N * HID);
    float* X      = (float*)alloc(sizeof(float) * (size_t)N * HID);
    float* emb    = (float*)alloc(sizeof(float) * (size_t)G * 2 * HID);
    float* counts = (float*)alloc(sizeof(float) * G);
    typedef unsigned short us;
    us* xbf   = (us*)alloc(sizeof(us) * (size_t)N * KPI);   // layer input bf16 (reused)
    us* w1t   = (us*)alloc(sizeof(us) * (size_t)HID * KPI);
    us* w2t   = (us*)alloc(sizeof(us) * (size_t)HID * HID);
    us* w3t   = (us*)alloc(sizeof(us) * (size_t)HID * HID);
    us* cw1t  = (us*)alloc(sizeof(us) * (size_t)HID * 2 * HID);
    us* cw2t  = (us*)alloc(sizeof(us) * (size_t)H2 * HID);
    us* cw3t  = (us*)alloc(sizeof(us) * (size_t)16 * H2);
    us* embbf = (us*)alloc(sizeof(us) * (size_t)G * 2 * HID);
    us* z1b   = (us*)alloc(sizeof(us) * (size_t)G * HID);
    us* z2b   = (us*)alloc(sizeof(us) * (size_t)G * H2);

    const int T = 256;
    const long long nh  = (long long)N * HID;
    const long long e16 = (long long)E * 16;
    const int gB = ((N >> 4) + 7) / 8;           // blocks for M=N GEMMs
    const int cB = ((G >> 4) + 7) / 8;           // blocks for M=G GEMMs

    // degrees (with self loop) and d^{-1/2}
    fill_kernel<<<(int)cdivll(N, T), T, 0, stream>>>(deg, 1.0f, N);
    deg_kernel<<<(int)cdivll(E, T), T, 0, stream>>>(dst, deg, E);
    rsqrt_kernel<<<(int)cdivll(N, T), T, 0, stream>>>(deg, dinv, N);

    // pack inputs / weights to padded bf16 (weights transposed)
    pack_a_kernel<<<(int)cdivll((long long)N * KPI, T), T, 0, stream>>>(x, xbf, N, IN, KPI);
    pack_bt_kernel<<<(int)cdivll((long long)HID * KPI, T), T, 0, stream>>>(W1, w1t, IN, HID, KPI, HID);
    pack_bt_kernel<<<(int)cdivll((long long)HID * HID, T), T, 0, stream>>>(W2, w2t, HID, HID, HID, HID);
    pack_bt_kernel<<<(int)cdivll((long long)HID * HID, T), T, 0, stream>>>(W3, w3t, HID, HID, HID, HID);
    pack_bt_kernel<<<(int)cdivll((long long)HID * 2 * HID, T), T, 0, stream>>>(cw1, cw1t, 2 * HID, HID, 2 * HID, HID);
    pack_bt_kernel<<<(int)cdivll((long long)H2 * HID, T), T, 0, stream>>>(cw2, cw2t, HID, H2, HID, H2);
    pack_bt_kernel<<<(int)cdivll((long long)16 * H2, T), T, 0, stream>>>(cw3, cw3t, H2, TGT, H2, 16);

    // 3 GCN layers: H = Xin@W (WMMA), AGG = scatter, X = relu(AGG + H/deg + b)
    const us*    Wt[3] = {w1t, w2t, w3t};
    const float* bl[3] = {b1, b2, b3};
    for (int l = 0; l < 3; ++l) {
        if (l == 0)
            wmma_gemm_kernel<64, 4, EPI_PLAIN><<<gB, 256, 0, stream>>>(
                xbf, Wt[l], nullptr, nullptr, nullptr, Hbuf, nullptr, N, HID);
        else
            wmma_gemm_kernel<64, 4, EPI_PLAIN><<<gB, 256, 0, stream>>>(
                xbf, Wt[l], nullptr, nullptr, nullptr, Hbuf, nullptr, N, HID);
        fill_kernel<<<(int)cdivll(nh, T), T, 0, stream>>>(AGG, 0.0f, nh);
        edge_agg_kernel<<<(int)cdivll(e16, T), T, 0, stream>>>(src, dst, dinv, Hbuf, AGG, e16);
        combine_kernel<<<(int)cdivll(nh, T), T, 0, stream>>>(Hbuf, AGG, deg, bl[l], X, xbf, nh);
    }

    // global mean/max pool -> emb [G x 128], then bf16 pack
    const long long g2h = (long long)G * 2 * HID;
    const long long gh  = (long long)G * HID;
    fill_kernel<<<(int)cdivll(g2h, T), T, 0, stream>>>(emb, 0.0f, g2h);
    fill_kernel<<<(int)cdivll(G, T), T, 0, stream>>>(counts, 0.0f, G);
    pool_kernel<<<(int)cdivll(nh, T), T, 0, stream>>>(X, batch, emb, counts, nh);
    pool_fin_kernel<<<(int)cdivll(gh, T), T, 0, stream>>>(emb, counts, gh);
    pack_a_kernel<<<(int)cdivll(g2h, T), T, 0, stream>>>(emb, embbf, G, 2 * HID, 2 * HID);

    // classifier MLP (bias + eval-BN + ReLU fused; bf16 chaining)
    wmma_gemm_kernel<128, 4, EPI_BNRELU><<<cB, 256, 0, stream>>>(
        embbf, cw1t, cb1, g1, be1, nullptr, z1b, G, HID);
    wmma_gemm_kernel<64, 2, EPI_BNRELU><<<cB, 256, 0, stream>>>(
        z1b, cw2t, cb2, g2, be2, nullptr, z2b, G, H2);
    wmma_gemm_kernel<32, 1, EPI_OUT><<<cB, 256, 0, stream>>>(
        z2b, cw3t, cb3, nullptr, nullptr, (float*)d_out, nullptr, G, TGT);
}
